// Custom_image_attention_38053410242587
// MI455X (gfx1250) — compile-verified, matrix operand order checked by reading, not run on hardware
//
#include <hip/hip_runtime.h>
#include <hip/hip_bf16.h>

typedef _Float16 v16h __attribute__((ext_vector_type(16)));
typedef _Float16 v8h  __attribute__((ext_vector_type(8)));
typedef float    v8f  __attribute__((ext_vector_type(8)));

#define BH      (8 * 16)
#define SEQ     1024
#define DH      64
#define DMODEL  1024
#define ROWS    8192   // B*N

__device__ __forceinline__ v8f wmma_f16(v16h a, v16h b, v8f c) {
    // D = A(16x32 f16) * B(32x16 f16) + C(16x16 f32)
    return __builtin_amdgcn_wmma_f32_16x16x32_f16(
        /*neg_a=*/false, a, /*neg_b=*/false, b,
        /*c_mod=*/(short)0, c, /*reuse_a=*/false, /*reuse_b=*/false);
}

// Load one f16 A/B fragment (16x32) given a row-major row pointer with the
// per-lane k-offsets: {8*half .. +7} and {16+8*half .. +7}.
__device__ __forceinline__ v16h frag_rowmajor(const _Float16* rowp, int half_) {
    v16h f;
    ((v8h*)&f)[0] = *(const v8h*)(rowp + half_ * 8);
    ((v8h*)&f)[1] = *(const v8h*)(rowp + half_ * 8 + 16);
    return f;
}

// ---------------------------------------------------------------------------
// Pre-pass A: x f32 -> xh f16 (row-major [8192][1024]); one-shot, bandwidth-
// bound (32 MB read / 16 MB write), removes all cvt work from the GEMM loop.
// ---------------------------------------------------------------------------
__global__ __launch_bounds__(256) void convert_x_kernel(
    const float* __restrict__ x, _Float16* __restrict__ xh)
{
    const size_t i = ((size_t)blockIdx.x * 256 + threadIdx.x) * 8;
    const float4* s4 = (const float4*)(x + i);
    float4 f0 = s4[0], f1 = s4[1];
    v8h h;
    h[0] = (_Float16)f0.x; h[1] = (_Float16)f0.y;
    h[2] = (_Float16)f0.z; h[3] = (_Float16)f0.w;
    h[4] = (_Float16)f1.x; h[5] = (_Float16)f1.y;
    h[6] = (_Float16)f1.z; h[7] = (_Float16)f1.w;
    *(v8h*)(xh + i) = h;
}

// ---------------------------------------------------------------------------
// Pre-pass B: W f32 [k][n] -> Wt f16 [n][k] (blockIdx.z selects Q/K/V weight).
// 32x32 LDS-tiled transpose.
// ---------------------------------------------------------------------------
__global__ __launch_bounds__(256) void transpose_w_kernel(
    const float* __restrict__ Wq, const float* __restrict__ Wk,
    const float* __restrict__ Wv, _Float16* __restrict__ wt)
{
    __shared__ float tile[32][33];
    const int which = blockIdx.z;
    const float* __restrict__ W = (which == 0) ? Wq : (which == 1) ? Wk : Wv;
    _Float16* __restrict__ Wt = wt + (size_t)which * DMODEL * DMODEL;

    const int n0 = blockIdx.x * 32;     // output-col block (rows of Wt)
    const int k0 = blockIdx.y * 32;     // k block
    const int tx = threadIdx.x & 31;
    const int ty = threadIdx.x >> 5;    // 0..7
    #pragma unroll
    for (int j = 0; j < 4; ++j) {
        const int r = ty * 4 + j;       // 0..31
        tile[r][tx] = W[(size_t)(k0 + r) * DMODEL + n0 + tx];
    }
    __syncthreads();
    #pragma unroll
    for (int j = 0; j < 4; ++j) {
        const int r = ty * 4 + j;
        Wt[(size_t)(n0 + r) * DMODEL + k0 + tx] = (_Float16)tile[tx][r];
    }
}

// ---------------------------------------------------------------------------
// Kernel 1: C = xh @ W (all f16 in, f16 out), W selected by blockIdx.z.
// 128x128 output tile per workgroup, 8 waves each computing a 2(M)x4(N) grid
// of 16x16 accumulators: 8 WMMAs per 6 fragment loads per k-chunk of 32.
// Q/K stored [bh][n][dh]; V stored transposed [bh][dh][n].
// ---------------------------------------------------------------------------
__global__ __launch_bounds__(256) void qkv_proj_kernel(
    const _Float16* __restrict__ xh, const _Float16* __restrict__ wt,
    _Float16* __restrict__ q_out, _Float16* __restrict__ k_out,
    _Float16* __restrict__ vt_out)
{
    __shared__ __align__(16) _Float16 Alds[128][32];   // [m][k]
    __shared__ __align__(16) _Float16 Blds[128][32];   // [n][k] (already transposed)

    const int which = blockIdx.z;
    const _Float16* __restrict__ Wtp = wt + (size_t)which * DMODEL * DMODEL;

    const int m0 = blockIdx.x * 128;    // token-row base (0..8191)
    const int n0 = blockIdx.y * 128;    // output-col base (0..1023)
    const int tid = threadIdx.x;
    const int lane = tid & 31;
    const int wave = tid >> 5;
    const int half_ = lane >> 4;
    const int l15 = lane & 15;

    const int mw = wave & 3;            // M-tile pair {2mw, 2mw+1}
    const int nw = wave >> 1 >> 1;      // N-tile quad {4nw .. 4nw+3}

    v8f acc[2][4] = {};

    // staging: thread copies one 32B strip of A and of B per chunk
    const int sr = tid >> 1;            // row 0..127
    const int sc = (tid & 1) * 16;      // col 0 or 16

    for (int kc = 0; kc < DMODEL; kc += 32) {
        const _Float16* ga = xh  + (size_t)(m0 + sr) * DMODEL + kc + sc;
        const _Float16* gb = Wtp + (size_t)(n0 + sr) * DMODEL + kc + sc;
        *(v8h*)&Alds[sr][sc]     = *(const v8h*)(ga);
        *(v8h*)&Alds[sr][sc + 8] = *(const v8h*)(ga + 8);
        *(v8h*)&Blds[sr][sc]     = *(const v8h*)(gb);
        *(v8h*)&Blds[sr][sc + 8] = *(const v8h*)(gb + 8);
        __syncthreads();

        v16h af[2], bf[4];
        #pragma unroll
        for (int i = 0; i < 2; ++i)
            af[i] = frag_rowmajor(&Alds[(mw * 2 + i) * 16 + l15][0], half_);
        #pragma unroll
        for (int j = 0; j < 4; ++j)
            bf[j] = frag_rowmajor(&Blds[(nw * 4 + j) * 16 + l15][0], half_);
        #pragma unroll
        for (int i = 0; i < 2; ++i)
            #pragma unroll
            for (int j = 0; j < 4; ++j)
                acc[i][j] = wmma_f16(af[i], bf[j], acc[i][j]);
        __syncthreads();
    }

    // --- write back: C element (M = g + 8*half, N = l15) ---
    #pragma unroll
    for (int i = 0; i < 2; ++i) {
        #pragma unroll
        for (int j = 0; j < 4; ++j) {
            const int col = n0 + (nw * 4 + j) * 16 + l15;
            const int h = col >> 6, d = col & 63;
            #pragma unroll
            for (int g = 0; g < 8; ++g) {
                const int mrow = m0 + (mw * 2 + i) * 16 + g + 8 * half_;
                const int bb = mrow >> 10, nn = mrow & 1023;
                const size_t bh = (size_t)bb * 16 + h;
                const _Float16 v = (_Float16)acc[i][j][g];
                if (which == 0)      q_out[(bh * SEQ + nn) * DH + d] = v;
                else if (which == 1) k_out[(bh * SEQ + nn) * DH + d] = v;
                else                 vt_out[(bh * DH + d) * SEQ + nn] = v;
            }
        }
    }
}

// ---------------------------------------------------------------------------
// Kernel 2: attention per (b, h, 32-query block). 8 waves.
//  Stage 1: S[32][1024] = Q_blk K^T into LDS (f32)
//  Stage 2: softmax rows in LDS, emit P f16
//  Stage 3: O = P V  (V transposed in ws), written f32 to d_out [B,N,H*dh]
// LDS: 128KB S + 64KB P + reduction scratch (~197KB dynamic; CDNA5 WGP has
// 320KB so a full 32x1024 score panel fits -> no online-softmax needed).
// ---------------------------------------------------------------------------
__global__ __launch_bounds__(256) void attn_kernel(
    const _Float16* __restrict__ Q, const _Float16* __restrict__ K,
    const _Float16* __restrict__ Vt, float* __restrict__ out)
{
    extern __shared__ __align__(16) char smem[];
    float*    S = (float*)smem;                                 // [32][1024]
    _Float16* P = (_Float16*)(smem + 32 * SEQ * 4);             // [32][1024]
    float*  red = (float*)(smem + 32 * SEQ * 4 + 32 * SEQ * 2); // [32][8]
    float* stat = red + 32 * 8;                                 // [32]

    const int q0 = blockIdx.x * 32;
    const size_t bh = (size_t)blockIdx.z * 16 + blockIdx.y;
    const _Float16* __restrict__ Qp = Q  + bh * SEQ * DH;
    const _Float16* __restrict__ Kp = K  + bh * SEQ * DH;
    const _Float16* __restrict__ Vp = Vt + bh * DH * SEQ;

    const int tid = threadIdx.x;
    const int lane = tid & 31;
    const int wave = tid >> 5;
    const int half_ = lane >> 4;
    const int l15 = lane & 15;

    // ---- Stage 1: S = Q K^T ----
    {
        const int qt   = wave & 1;            // query tile (0/1)
        const int key0 = (wave >> 1) * 256;   // 256 keys per wave
        const _Float16* qrow = Qp + (size_t)(q0 + qt * 16 + l15) * DH;
        const v16h a0 = frag_rowmajor(qrow, half_);        // d = 0..31
        const v16h a1 = frag_rowmajor(qrow + 32, half_);   // d = 32..63
        for (int kt = 0; kt < 16; ++kt) {
            const int m0 = key0 + kt * 16;
            const _Float16* krow = Kp + (size_t)(m0 + l15) * DH;
            if (kt + 1 < 16)   // warm L0/L2 for next key tile
                __builtin_prefetch(krow + 16 * DH, 0, 0);
            const v16h b0 = frag_rowmajor(krow, half_);
            const v16h b1 = frag_rowmajor(krow + 32, half_);
            v8f acc = {};
            acc = wmma_f16(a0, b0, acc);
            acc = wmma_f16(a1, b1, acc);
            #pragma unroll
            for (int g = 0; g < 8; ++g)
                S[(qt * 16 + g + 8 * half_) * SEQ + m0 + l15] = acc[g];
        }
    }
    __syncthreads();

    // ---- Stage 2: softmax over each of 32 rows (8 threads/row) ----
    {
        const int r = tid >> 3, i8 = tid & 7;
        float* Sr = S + (size_t)r * SEQ;
        const int j0 = i8 * 128, j1 = j0 + 128;
        float m = -3.0e38f;
        for (int j = j0; j < j1; ++j) m = fmaxf(m, Sr[j]);
        red[r * 8 + i8] = m;
        __syncthreads();
        if (i8 == 0) {
            float mm = red[r * 8];
            #pragma unroll
            for (int j = 1; j < 8; ++j) mm = fmaxf(mm, red[r * 8 + j]);
            stat[r] = mm;
        }
        __syncthreads();
        const float mm = stat[r];
        float s = 0.f;
        for (int j = j0; j < j1; ++j) { float e = __expf(Sr[j] - mm); Sr[j] = e; s += e; }
        __syncthreads();
        red[r * 8 + i8] = s;
        __syncthreads();
        if (i8 == 0) {
            float ss = 0.f;
            #pragma unroll
            for (int j = 0; j < 8; ++j) ss += red[r * 8 + j];
            stat[r] = ss;
        }
        __syncthreads();
        const float inv = 1.0f / stat[r];
        _Float16* Pr = P + (size_t)r * SEQ;
        for (int j = j0; j < j1; ++j) Pr[j] = (_Float16)(Sr[j] * inv);
    }
    __syncthreads();

    // ---- Stage 3: O = P V  (each wave: one 16x16 output tile) ----
    {
        const int qt = wave & 1;      // query tile
        const int dt = wave >> 1;     // dh tile (0..3)
        v8f acc = {};
        const _Float16* prow = P  + (size_t)(qt * 16 + l15) * SEQ;
        const _Float16* vrow = Vp + (size_t)(dt * 16 + l15) * SEQ;
        for (int kc = 0; kc < SEQ; kc += 32) {
            const v16h a = frag_rowmajor(prow + kc, half_);
            const v16h b = frag_rowmajor(vrow + kc, half_);
            acc = wmma_f16(a, b, acc);
        }
        const int d = (int)blockIdx.y * DH + dt * 16 + l15;
        #pragma unroll
        for (int g = 0; g < 8; ++g) {
            const int tok = q0 + qt * 16 + g + 8 * half_;
            out[((size_t)blockIdx.z * SEQ + tok) * DMODEL + d] = acc[g];
        }
    }
}

extern "C" void kernel_launch(void* const* d_in, const int* in_sizes, int n_in,
                              void* d_out, int out_size, void* d_ws, size_t ws_size,
                              hipStream_t stream) {
    const float* x  = (const float*)d_in[0];
    const float* Wq = (const float*)d_in[1];
    const float* Wk = (const float*)d_in[2];
    const float* Wv = (const float*)d_in[3];
    float* out = (float*)d_out;

    const size_t per = (size_t)BH * SEQ * DH;   // 8M f16 elements
    _Float16* qws = (_Float16*)d_ws;
    _Float16* kws = qws + per;
    _Float16* vws = kws + per;
    _Float16* xh  = vws + per;                          // [8192][1024] f16
    _Float16* wt  = xh + (size_t)ROWS * DMODEL;         // 3x[1024][1024] f16 (transposed)

    // Pre-pass: f32 -> f16 (+ W transpose); one-shot, bandwidth-trivial.
    convert_x_kernel<<<(ROWS * DMODEL) / (256 * 8), 256, 0, stream>>>(x, xh);
    dim3 gt(DMODEL / 32, DMODEL / 32, 3);
    transpose_w_kernel<<<gt, 256, 0, stream>>>(Wq, Wk, Wv, wt);

    // QKV projection GEMMs (128x128 tiles).
    dim3 g1(ROWS / 128, DMODEL / 128, 3);
    qkv_proj_kernel<<<g1, 256, 0, stream>>>(xh, wt, qws, kws, vws);

    // Attention.
    const size_t shmem = 32 * SEQ * 4   // S f32
                       + 32 * SEQ * 2   // P f16
                       + (32 * 8 + 32) * 4;   // reduction scratch
    hipFuncSetAttribute((const void*)attn_kernel,
                        hipFuncAttributeMaxDynamicSharedMemorySize, (int)shmem);
    dim3 g2(SEQ / 32, 16, 8);
    attn_kernel<<<g2, 256, shmem, stream>>>(qws, kws, vws, out);
}
